// BahdanauAttention_21552145891360
// MI455X (gfx1250) — matrix operand order repaired and checked by reading, not compiled, not run on hardware
//
#include <hip/hip_runtime.h>

#define H  1024
#define S  2048
#define BB 32

typedef __bf16 v16bf __attribute__((ext_vector_type(16)));
typedef __bf16 v8bf  __attribute__((ext_vector_type(8)));
typedef __bf16 v4bf  __attribute__((ext_vector_type(4)));
typedef float  v8f   __attribute__((ext_vector_type(8)));

// Branch-free tanh: tanh(x) = 1 - 2/(exp(2x)+1).
// exp(2x)->inf => rcp->0 => +1 ; exp(2x)->0 => 1-2 => -1  (no branches)
__device__ __forceinline__ float fast_tanh(float x) {
  float e = __expf(2.0f * x);                 // v_mul + v_exp_f32
  float r = __builtin_amdgcn_rcpf(e + 1.0f);  // v_rcp_f32
  return __builtin_fmaf(-2.0f, r, 1.0f);      // v_fma_f32
}

// ---------------------------------------------------------------------------
// W_h (HxH f32 row-major) -> Wt bf16 with Wt[n][k] = W_h[k][n], so a B-matrix
// fragment (column n, 16 consecutive K values) is one contiguous 32B load.
// ---------------------------------------------------------------------------
__global__ void k_transpose_wh(const float* __restrict__ Wh,
                               unsigned short* __restrict__ Wt_) {
  __bf16* Wt = (__bf16*)Wt_;
  __shared__ float tile[32][33];
  int n0 = blockIdx.x * 32, k0 = blockIdx.y * 32;
  int tx = threadIdx.x, ty = threadIdx.y;  // block (32, 8)
#pragma unroll
  for (int j = 0; j < 4; ++j)
    tile[ty + 8 * j][tx] = Wh[(size_t)(k0 + ty + 8 * j) * H + (n0 + tx)];
  __syncthreads();
#pragma unroll
  for (int j = 0; j < 4; ++j)
    Wt[(size_t)(n0 + ty + 8 * j) * H + (k0 + tx)] = (__bf16)tile[tx][ty + 8 * j];
}

// ---------------------------------------------------------------------------
// proj_s[b][d] = sum_h query[b][h] * W_s[h][d]   (tiny, f32)
// ---------------------------------------------------------------------------
__global__ void k_proj_s(const float* __restrict__ q,
                         const float* __restrict__ Ws,
                         float* __restrict__ projs) {
  int b = blockIdx.x;
  int d = blockIdx.y * 256 + threadIdx.x;
  float acc = 0.f;
  for (int h = 0; h < H; ++h) acc += q[(size_t)b * H + h] * Ws[(size_t)h * H + d];
  projs[(size_t)b * H + d] = acc;
}

// ---------------------------------------------------------------------------
// Fused GEMM + tanh + v-dot:
//   scores[m] = sum_n v[n] * tanh( (enc[m,:] @ W_h)[n] + proj_s[b][n] )
// One wave per 32-row M tile (two 16-row WMMA tiles). Each B fragment loaded
// from L2 once and reused by two v_wmma_f32_16x16x32_bf16 -> halves L2 traffic
// per WMMA (1x global_load_b128 + 2x ds_load_b128 per matrix op).
// ---------------------------------------------------------------------------
__global__ __launch_bounds__(32)
void k_scores(const float* __restrict__ enc,
              const unsigned short* __restrict__ Wt_,
              const float* __restrict__ projs,
              const float* __restrict__ vvec,
              float* __restrict__ scores) {
  const __bf16* Wt = (const __bf16*)Wt_;
  __shared__ __bf16 As[32 * H];  // 64 KB: 32-row A tile in bf16, row-major

  const int lane = threadIdx.x;
  const int m0   = blockIdx.x * 32;  // global row = b*S + s; tile stays in one b
  const int b    = m0 >> 11;         // m0 / S

  // Stage + convert A tile: 32 x 1024 f32 -> bf16 LDS (coalesced float4 loads)
  const float4* Ap = (const float4*)(enc + (size_t)m0 * H);
  for (int t = lane; t < 32 * H / 4; t += 32) {
    float4 f = Ap[t];
    v4bf o;
    o[0] = (__bf16)f.x; o[1] = (__bf16)f.y; o[2] = (__bf16)f.z; o[3] = (__bf16)f.w;
    ((v4bf*)As)[t] = o;
  }
  __syncthreads();

  const int half = lane >> 4;   // 0: lanes 0-15, 1: lanes 16-31
  const int r    = lane & 15;

  float sacc0[8], sacc1[8];
#pragma unroll
  for (int i = 0; i < 8; ++i) { sacc0[i] = 0.f; sacc1[i] = 0.f; }

  for (int n0 = 0; n0 < H; n0 += 16) {
    const int   nn   = n0 + r;                 // this lane's output column
    const float bias = projs[(size_t)b * H + nn];
    const float vn   = vvec[nn];

    v8f acc0 = {}, acc1 = {};
    const v8bf*   Arow0 = (const v8bf*)(As + r * H);         // rows m0..m0+15
    const v8bf*   Arow1 = (const v8bf*)(As + (16 + r) * H);  // rows m0+16..+31
    const __bf16* Bcol  = Wt + (size_t)nn * H + half * 16;

#pragma unroll 4
    for (int k0 = 0; k0 < H; k0 += 32) {
      // B (32x16), contiguous k order used here: N = lane&15, lanes 0-15 K k0..k0+15, lanes 16-31
      // K k0+16..k0+31; packed Wt makes this one contiguous 32B load.
      v16bf bfrag = *(const v16bf*)(Bcol + k0);

      // A (16x32), contiguous k order used here: lanes 0-15 hold K k0..k0+15,
      // lanes 16-31 hold K k0+16..k0+31 (row = lane&15), matching B.
      const int ilo = (k0 + half * 16) >> 3;
      const int ihi = (k0 + half * 16 + 8) >> 3;
      v8bf a0lo = Arow0[ilo], a0hi = Arow0[ihi];
      v8bf a1lo = Arow1[ilo], a1hi = Arow1[ihi];
      v16bf a0, a1;
#pragma unroll
      for (int j = 0; j < 8; ++j) {
        a0[j] = a0lo[j]; a0[j + 8] = a0hi[j];
        a1[j] = a1lo[j]; a1[j + 8] = a1hi[j];
      }

      acc0 = __builtin_amdgcn_wmma_f32_16x16x32_bf16(
          false, a0, false, bfrag, (short)0, acc0, false, false);
      acc1 = __builtin_amdgcn_wmma_f32_16x16x32_bf16(
          false, a1, false, bfrag, (short)0, acc1, false, false);
    }

    // C/D layout: lane holds column nn, rows i + half*8 in acc[i].
    // Branch-free tanh keeps EXEC untouched between WMMA bursts.
#pragma unroll
    for (int i = 0; i < 8; ++i) {
      sacc0[i] += fast_tanh(acc0[i] + bias) * vn;
      sacc1[i] += fast_tanh(acc1[i] + bias) * vn;
    }
  }

  // Reduce over the 16 columns each half-wave holds
#pragma unroll
  for (int i = 0; i < 8; ++i) {
    float s0 = sacc0[i], s1 = sacc1[i];
    s0 += __shfl_xor(s0, 1, 16); s1 += __shfl_xor(s1, 1, 16);
    s0 += __shfl_xor(s0, 2, 16); s1 += __shfl_xor(s1, 2, 16);
    s0 += __shfl_xor(s0, 4, 16); s1 += __shfl_xor(s1, 4, 16);
    s0 += __shfl_xor(s0, 8, 16); s1 += __shfl_xor(s1, 8, 16);
    sacc0[i] = s0; sacc1[i] = s1;
  }
  if (r == 0) {
#pragma unroll
    for (int i = 0; i < 8; ++i) {
      scores[m0 + half * 8 + i]      = sacc0[i];
      scores[m0 + 16 + half * 8 + i] = sacc1[i];
    }
  }
}

// ---------------------------------------------------------------------------
// Masked softmax over S per batch (mask via -1e9, matching the reference)
// ---------------------------------------------------------------------------
__global__ void k_softmax(const float* __restrict__ scores,
                          const long long* __restrict__ lens,
                          float* __restrict__ wts) {
  __shared__ float red[256];
  int b = blockIdx.x, tid = threadIdx.x;
  long long L = lens[b];

  float ls[8];
  float m = -3.0e38f;
#pragma unroll
  for (int j = 0; j < 8; ++j) {
    int s = tid + j * 256;
    float sc = scores[(size_t)b * S + s];
    sc = ((long long)s < L) ? sc : -1e9f;
    ls[j] = sc;
    m = fmaxf(m, sc);
  }
  red[tid] = m; __syncthreads();
  for (int o = 128; o > 0; o >>= 1) {
    if (tid < o) red[tid] = fmaxf(red[tid], red[tid + o]);
    __syncthreads();
  }
  m = red[0]; __syncthreads();

  float le[8], sum = 0.f;
#pragma unroll
  for (int j = 0; j < 8; ++j) { le[j] = __expf(ls[j] - m); sum += le[j]; }
  red[tid] = sum; __syncthreads();
  for (int o = 128; o > 0; o >>= 1) {
    if (tid < o) red[tid] += red[tid + o];
    __syncthreads();
  }
  float inv = 1.0f / red[0];
#pragma unroll
  for (int j = 0; j < 8; ++j) wts[(size_t)b * S + tid + j * 256] = le[j] * inv;
}

// ---------------------------------------------------------------------------
// attn_out[b][h] = sum_s w[b][s] * enc[b][s][h]  — split-S partials (no atomics)
// ---------------------------------------------------------------------------
__global__ void k_wsum_partial(const float* __restrict__ enc,
                               const float* __restrict__ wts,
                               float* __restrict__ partial) {
  int b = blockIdx.x, cy = blockIdx.y, tid = threadIdx.x;
  int s0 = cy * 128;
  float4 acc = {0.f, 0.f, 0.f, 0.f};
  const float4* ep = (const float4*)(enc + ((size_t)b * S + s0) * H);
  for (int s = 0; s < 128; ++s) {
    float w = wts[(size_t)b * S + s0 + s];
    float4 e = ep[(size_t)s * (H / 4) + tid];
    acc.x += w * e.x; acc.y += w * e.y; acc.z += w * e.z; acc.w += w * e.w;
  }
  ((float4*)partial)[((size_t)cy * BB + b) * (H / 4) + tid] = acc;
}

__global__ void k_wsum_final(const float* __restrict__ partial,
                             float* __restrict__ out) {
  int b = blockIdx.x, tid = threadIdx.x;
  float4 acc = {0.f, 0.f, 0.f, 0.f};
  for (int c = 0; c < S / 128; ++c) {
    float4 p = ((const float4*)partial)[((size_t)c * BB + b) * (H / 4) + tid];
    acc.x += p.x; acc.y += p.y; acc.z += p.z; acc.w += p.w;
  }
  ((float4*)out)[(size_t)b * (H / 4) + tid] = acc;
}

// ---------------------------------------------------------------------------
extern "C" void kernel_launch(void* const* d_in, const int* in_sizes, int n_in,
                              void* d_out, int out_size, void* d_ws, size_t ws_size,
                              hipStream_t stream) {
  const float*     query = (const float*)d_in[0];      // (B,1,H)
  const float*     enc   = (const float*)d_in[1];      // (B,S,H)
  const long long* lens  = (const long long*)d_in[2];  // (B,) int64
  const float*     Wh    = (const float*)d_in[3];      // (H,H)
  const float*     Ws    = (const float*)d_in[4];      // (H,H)
  const float*     vv    = (const float*)d_in[5];      // (H,)
  float* out = (float*)d_out;                          // (B,1,H)

  char* ws = (char*)d_ws;
  unsigned short* Wt  = (unsigned short*)ws;                        // 2 MB bf16
  float* projs   = (float*)(ws + (2u << 20));                       // 128 KB
  float* scores  = (float*)(ws + (2u << 20) + (128u << 10));        // 256 KB
  float* wts     = (float*)(ws + (2u << 20) + (384u << 10));        // 256 KB
  float* partial = (float*)(ws + (2u << 20) + (640u << 10));        // 2 MB

  k_transpose_wh<<<dim3(H / 32, H / 32), dim3(32, 8), 0, stream>>>(Wh, Wt);
  k_proj_s     <<<dim3(BB, H / 256), 256, 0, stream>>>(query, Ws, projs);
  k_scores     <<<dim3(BB * S / 32), 32, 0, stream>>>(enc, Wt, projs, vv, scores);
  k_softmax    <<<dim3(BB), 256, 0, stream>>>(scores, lens, wts);
  k_wsum_partial<<<dim3(BB, S / 128), 256, 0, stream>>>(enc, wts, partial);
  k_wsum_final <<<dim3(BB), 256, 0, stream>>>(partial, out);
}
